// GRU_Repeat_AutoEncoder_29334626632265
// MI455X (gfx1250) — matrix operand order repaired and checked by reading, not compiled, not run on hardware
//
#include <hip/hip_runtime.h>

// ---------------------------------------------------------------------------
// GRU Repeat AutoEncoder, fused for MI455X (gfx1250, wave32, WMMA bf16).
// Serial-latency-bound: 512-step scans run inside persistent per-batch-tile
// workgroups (16 rows x 16 WGs). Loop-invariant weights are staged into LDS
// (dynamic LDS, up to ~308KB of the 320KB WGP budget) via async-to-LDS copies
// so each serial step reads B-panels at LDS latency instead of L2 latency.
// ---------------------------------------------------------------------------

typedef unsigned short u16;
typedef __attribute__((ext_vector_type(16))) __bf16 v16bf;
typedef __attribute__((ext_vector_type(8)))  float  v8f;
typedef __attribute__((ext_vector_type(4)))  int    v4i;

static constexpr int Bc  = 256;
static constexpr int Sc  = 512;
static constexpr int Dc  = 128;
static constexpr int Hc  = 256;
static constexpr int HHc = 128;
static constexpr int Lc  = 64;
static constexpr int PAD = 8;      // u16/f32 row padding to dodge bank conflicts

// ---- CDNA5 async global->LDS staging (guarded; falls back to sync copy) ----
#if defined(__has_builtin)
#if __has_builtin(__builtin_amdgcn_global_load_async_to_lds_b128) && \
    __has_builtin(__builtin_amdgcn_s_wait_asynccnt)
#define ASYNC_LDS 1
#endif
#endif
#ifndef ASYNC_LDS
#define ASYNC_LDS 0
#endif

__device__ __forceinline__ void stage16(const u16* g, u16* l) {
#if ASYNC_LDS
    __builtin_amdgcn_global_load_async_to_lds_b128(
        (__attribute__((address_space(1))) v4i*)(u16*)g,
        (__attribute__((address_space(3))) v4i*)l, 0, 0);
#else
    *(uint4*)l = *(const uint4*)g;
#endif
}

__device__ __forceinline__ void stage_wait_and_sync() {
#if ASYNC_LDS
    __builtin_amdgcn_s_wait_asynccnt(0);
#endif
    __syncthreads();
}

// cooperative copy of n16*8 u16 elements (16B granules) global -> LDS
__device__ __forceinline__ void stage_matrix(const u16* g, u16* l, int n16,
                                             int tid, int nthreads) {
    for (int i = tid; i < n16; i += nthreads)
        stage16(g + (size_t)i * 8, l + (size_t)i * 8);
}

// ---------------------------------------------------------------------------
union AF {            // one WMMA bf16 operand fragment (16 bf16 per lane)
    v16bf v;
    uint4 q[2];
};

__device__ __forceinline__ u16 f2bf(float x) {
    __bf16 h = (__bf16)x;                 // native v_cvt path on gfx1250
    return __builtin_bit_cast(u16, h);
}

__device__ __forceinline__ float sigmoidf(float x) {
    return 1.0f / (1.0f + __expf(-x));
}

__device__ __forceinline__ v8f wmma_bf16(const AF& a, const AF& b, v8f c) {
    return __builtin_amdgcn_wmma_f32_16x16x32_bf16(
        false, a.v, false, b.v, (short)0, c, false, false);
}

// A fragment: rows 0..15 (lane&15 picks row); caller pre-offsets the row ptr
// by row*ldk + 8*half; kb selects the 32-wide K block.
__device__ __forceinline__ AF load_a_ptr(const u16* row_ptr_half, int kb) {
    AF a;
    a.q[0] = *(const uint4*)(row_ptr_half + kb);
    a.q[1] = *(const uint4*)(row_ptr_half + kb + 16);
    return a;
}

// B fragment from row-major W[Nout, K] (computing X @ W^T): output col n0+c,
// 8 bf16 along K at kb + 8*half and 8 at kb + 16 + 8*half. Works for global or LDS W.
__device__ __forceinline__ AF load_b(const u16* W, int n0, int ldk, int kb,
                                     int c, int half) {
    AF b;
    const u16* p = W + (size_t)(n0 + c) * ldk + kb + 8 * half;
    b.q[0] = *(const uint4*)(p);
    b.q[1] = *(const uint4*)(p + 16);
    return b;
}

// ---------------------------------------------------------------------------
__global__ void zero_loss_kernel(float* p) {
    if (threadIdx.x == 0 && blockIdx.x == 0) p[0] = 0.0f;
}

__global__ void convert_bf16_kernel(const float* __restrict__ src,
                                    u16* __restrict__ dst, int n) {
    int i = blockIdx.x * blockDim.x + threadIdx.x;
    if (i < n) dst[i] = f2bf(src[i]);
}

__global__ void embed_kernel(const int* __restrict__ tokens,
                             const float* __restrict__ path_w,
                             const float* __restrict__ sys_w,
                             float* __restrict__ emb_f,
                             u16* __restrict__ emb_bf) {
    int idx = blockIdx.x * 256 + threadIdx.x;   // over B*S*D
    int bs = idx >> 7;
    int d  = idx & 127;
    int t0 = tokens[bs * 2 + 0];
    int t1 = tokens[bs * 2 + 1];
    float v = path_w[(size_t)t0 * Dc + d] + sys_w[(size_t)t1 * Dc + d];
    emb_f[idx]  = v;
    emb_bf[idx] = f2bf(v);
}

// ---------------------------------------------------------------------------
// Fused sequential GRU layer; weights optionally staged into dynamic LDS.
// ---------------------------------------------------------------------------
constexpr size_t gru_smem_bytes(int HID, int KIN, bool swih, bool swhh) {
    size_t s = (size_t)16 * (HID + PAD) * 4 + (size_t)16 * (HID + PAD) * 2;
    if (swih) s += (size_t)3 * HID * KIN * 2;
    if (swhh) s += (size_t)3 * HID * HID * 2;
    return s;
}

template <int HID, int KIN, bool SWIH, bool SWHH>
__global__ __launch_bounds__(256) void gru_layer_kernel(
    const u16* __restrict__ x_seq,    // [B,S,KIN] bf16
    const u16* __restrict__ Wih,      // [3*HID, KIN] bf16
    const u16* __restrict__ Whh,      // [3*HID, HID] bf16
    const float* __restrict__ bih,
    const float* __restrict__ bhh,
    u16* __restrict__ out_seq)        // [B,S,HID] bf16
{
    constexpr int NF  = HID / 128;
    constexpr int CPW = HID / 8;
    constexpr int LDH = HID + PAD;
    constexpr size_t OFF_HBF = (size_t)16 * LDH * 4;
    constexpr size_t OFF_W0  = OFF_HBF + (size_t)16 * LDH * 2;

    extern __shared__ __align__(16) char smem[];
    float* h_f  = (float*)smem;                    // [16][LDH]
    u16*   h_bf = (u16*)(smem + OFF_HBF);          // [16][LDH]

    const int tid  = threadIdx.x;
    const int wave = tid >> 5;
    const int lane = tid & 31;
    const int r    = lane & 15;
    const int half = lane >> 4;
    const int b0   = blockIdx.x * 16;
    const int cw0  = wave * CPW;

    for (int i = tid; i < 16 * LDH; i += 256) { h_f[i] = 0.0f; h_bf[i] = 0; }

    const u16* WihB = Wih;
    const u16* WhhB = Whh;
    if constexpr (SWIH) {
        u16* wl = (u16*)(smem + OFF_W0);
        stage_matrix(Wih, wl, 3 * HID * KIN / 8, tid, 256);
        WihB = wl;
    }
    if constexpr (SWHH) {
        u16* wl = (u16*)(smem + OFF_W0 + (SWIH ? (size_t)3 * HID * KIN * 2 : 0));
        stage_matrix(Whh, wl, 3 * HID * HID / 8, tid, 256);
        WhhB = wl;
    }
    stage_wait_and_sync();

    float bR[NF], bZ[NF], bNi[NF], bNh[NF];
#pragma unroll
    for (int f = 0; f < NF; ++f) {
        int col = cw0 + f * 16 + r;
        bR[f]  = bih[col] + bhh[col];
        bZ[f]  = bih[HID + col] + bhh[HID + col];
        bNi[f] = bih[2 * HID + col];
        bNh[f] = bhh[2 * HID + col];
    }

    const u16* xrow = x_seq + (size_t)(b0 + r) * Sc * KIN + 8 * half;

    for (int t = 0; t < Sc; ++t) {
        v8f accR[NF], accZ[NF], accNi[NF], accNh[NF];
#pragma unroll
        for (int f = 0; f < NF; ++f)
#pragma unroll
            for (int j = 0; j < 8; ++j) {
                accR[f][j] = bR[f]; accZ[f][j] = bZ[f];
                accNi[f][j] = bNi[f]; accNh[f][j] = bNh[f];
            }

        // ---- input contribution: x_t @ Wih^T ------------------------------
        const u16* xa = xrow + (size_t)t * KIN;
#pragma unroll
        for (int kb = 0; kb < KIN; kb += 32) {
            AF a = load_a_ptr(xa, kb);
#pragma unroll
            for (int f = 0; f < NF; ++f) {
                int n0 = cw0 + f * 16;
                accR[f]  = wmma_bf16(a, load_b(WihB, n0,           KIN, kb, r, half), accR[f]);
                accZ[f]  = wmma_bf16(a, load_b(WihB, HID + n0,     KIN, kb, r, half), accZ[f]);
                accNi[f] = wmma_bf16(a, load_b(WihB, 2 * HID + n0, KIN, kb, r, half), accNi[f]);
            }
        }
        // ---- hidden contribution: h_{t-1} @ Whh^T -------------------------
#pragma unroll
        for (int kb = 0; kb < HID; kb += 32) {
            AF a = load_a_ptr(&h_bf[r * LDH + 8 * half], kb);
#pragma unroll
            for (int f = 0; f < NF; ++f) {
                int n0 = cw0 + f * 16;
                accR[f]  = wmma_bf16(a, load_b(WhhB, n0,           HID, kb, r, half), accR[f]);
                accZ[f]  = wmma_bf16(a, load_b(WhhB, HID + n0,     HID, kb, r, half), accZ[f]);
                accNh[f] = wmma_bf16(a, load_b(WhhB, 2 * HID + n0, HID, kb, r, half), accNh[f]);
            }
        }
        __syncthreads();   // all waves done reading h_bf

#pragma unroll
        for (int f = 0; f < NF; ++f) {
            int col = cw0 + f * 16 + r;
#pragma unroll
            for (int j = 0; j < 8; ++j) {
                int m = half * 8 + j;
                float rg = sigmoidf(accR[f][j]);
                float zg = sigmoidf(accZ[f][j]);
                float ng = tanhf(accNi[f][j] + rg * accNh[f][j]);
                float ho = h_f[m * LDH + col];
                float hn = (1.0f - zg) * ng + zg * ho;
                h_f[m * LDH + col] = hn;
                u16 hb = f2bf(hn);
                h_bf[m * LDH + col] = hb;
                out_seq[((size_t)(b0 + m) * Sc + t) * HID + col] = hb;
            }
        }
        __syncthreads();
    }
}

// ---------------------------------------------------------------------------
__global__ __launch_bounds__(128) void latent_kernel(
    const u16* __restrict__ h2_seq,   // [B,S,HH] bf16
    const u16* __restrict__ l1W,      // [L, HH] bf16
    const float* __restrict__ l1_b,
    float* __restrict__ latent_out,   // [B, L] f32 (d_out)
    u16* __restrict__ latent_bf)      // [B, L] bf16
{
    const int tid  = threadIdx.x;
    const int wave = tid >> 5;
    const int lane = tid & 31;
    const int r    = lane & 15;
    const int half = lane >> 4;
    const int m0   = blockIdx.x * 16;
    const int n0   = wave * 16;

    v8f acc;
    float bv = l1_b[n0 + r];
#pragma unroll
    for (int j = 0; j < 8; ++j) acc[j] = bv;

    const u16* arow =
        h2_seq + ((size_t)(m0 + r) * Sc + (Sc - 1)) * HHc + 8 * half;
#pragma unroll
    for (int kb = 0; kb < HHc; kb += 32) {
        AF a = load_a_ptr(arow, kb);
        acc = wmma_bf16(a, load_b(l1W, n0, HHc, kb, r, half), acc);
    }
#pragma unroll
    for (int j = 0; j < 8; ++j) {
        int m = half * 8 + j;
        int col = n0 + r;
        float v = acc[j];
        latent_out[(size_t)(m0 + m) * Lc + col] = v;
        latent_bf [(size_t)(m0 + m) * Lc + col] = f2bf(v);
    }
}

// ---------------------------------------------------------------------------
// Fused repeat-decoder: cell1 (HH) -> cell2 (H, input GEMM fused) -> out
// linear + MSE, 512 steps. d1_Whh and l2_W staged to LDS; d2_* streamed from
// L2 with prefetch hints issued during cell1 so the fetch overlaps compute.
// ---------------------------------------------------------------------------
constexpr size_t DEC_LDH1  = HHc + PAD;
constexpr size_t DEC_LDH2  = Hc + PAD;
constexpr size_t DEC_LDG   = 3 * HHc + PAD;
constexpr size_t DEC_OFF_H1BF = 16 * DEC_LDH1 * 4;
constexpr size_t DEC_OFF_H2F  = DEC_OFF_H1BF + 16 * DEC_LDH1 * 2;
constexpr size_t DEC_OFF_H2BF = DEC_OFF_H2F + 16 * DEC_LDH2 * 4;
constexpr size_t DEC_OFF_GI1  = DEC_OFF_H2BF + 16 * DEC_LDH2 * 2;
constexpr size_t DEC_OFF_W1   = DEC_OFF_GI1 + 16 * DEC_LDG * 4;
constexpr size_t DEC_OFF_W2   = DEC_OFF_W1 + (size_t)3 * HHc * HHc * 2;
constexpr size_t DEC_SMEM     = DEC_OFF_W2 + (size_t)Dc * Hc * 2;

__global__ __launch_bounds__(256) void decoder_kernel(
    const u16* __restrict__ latent_bf,               // [B, L]
    const u16* __restrict__ d1Wih,                   // [3*HH, L]
    const u16* __restrict__ d1Whh,                   // [3*HH, HH]
    const float* __restrict__ d1_bih, const float* __restrict__ d1_bhh,
    const u16* __restrict__ d2Wih,                   // [3*H, HH]
    const u16* __restrict__ d2Whh,                   // [3*H, H]
    const float* __restrict__ d2_bih, const float* __restrict__ d2_bhh,
    const u16* __restrict__ l2W,                     // [D, H]
    const float* __restrict__ l2_b,
    const float* __restrict__ emb,                   // [B,S,D] f32
    float* __restrict__ loss_out)
{
    extern __shared__ __align__(16) char smem[];
    float* h1_f  = (float*)smem;                        // [16][DEC_LDH1]
    u16*   h1_bf = (u16*)(smem + DEC_OFF_H1BF);
    float* h2_f  = (float*)(smem + DEC_OFF_H2F);        // [16][DEC_LDH2]
    u16*   h2_bf = (u16*)(smem + DEC_OFF_H2BF);
    float* gi1   = (float*)(smem + DEC_OFF_GI1);        // [16][DEC_LDG]
    u16*   w1L   = (u16*)(smem + DEC_OFF_W1);           // d1Whh staged
    u16*   w2L   = (u16*)(smem + DEC_OFF_W2);           // l2W staged
    __shared__ float redbuf[8];

    const int tid  = threadIdx.x;
    const int wave = tid >> 5;
    const int lane = tid & 31;
    const int r    = lane & 15;
    const int half = lane >> 4;
    const int b0   = blockIdx.x * 16;

    for (int i = tid; i < 16 * (int)DEC_LDH1; i += 256) { h1_f[i] = 0.0f; h1_bf[i] = 0; }
    for (int i = tid; i < 16 * (int)DEC_LDH2; i += 256) { h2_f[i] = 0.0f; h2_bf[i] = 0; }
    stage_matrix(d1Whh, w1L, 3 * HHc * HHc / 8, tid, 256);
    stage_matrix(l2W,   w2L, Dc * Hc / 8,       tid, 256);
    stage_wait_and_sync();

    // ---- gi1 (constant across steps): wave covers 48 cols = 3 fragments ---
    {
        v8f acc[3];
#pragma unroll
        for (int f = 0; f < 3; ++f) {
            float bv = d1_bih[wave * 48 + f * 16 + r];
#pragma unroll
            for (int j = 0; j < 8; ++j) acc[f][j] = bv;
        }
        const u16* arow = latent_bf + (size_t)(b0 + r) * Lc + 8 * half;
#pragma unroll
        for (int kb = 0; kb < Lc; kb += 32) {
            AF a = load_a_ptr(arow, kb);
#pragma unroll
            for (int f = 0; f < 3; ++f)
                acc[f] = wmma_bf16(a, load_b(d1Wih, wave * 48 + f * 16, Lc, kb, r, half), acc[f]);
        }
#pragma unroll
        for (int f = 0; f < 3; ++f)
#pragma unroll
            for (int j = 0; j < 8; ++j)
                gi1[(half * 8 + j) * DEC_LDG + wave * 48 + f * 16 + r] = acc[f][j];
    }
    __syncthreads();

    const int c1 = wave * 16 + r;                 // cell1 column (HH)
    float b1R = d1_bhh[c1];
    float b1Z = d1_bhh[HHc + c1];
    float b1N = d1_bhh[2 * HHc + c1];
    float b2R[2], b2Z[2], b2Ni[2], b2Nh[2];
#pragma unroll
    for (int f = 0; f < 2; ++f) {
        int col = wave * 32 + f * 16 + r;         // cell2 column (H)
        b2R[f]  = d2_bih[col] + d2_bhh[col];
        b2Z[f]  = d2_bih[Hc + col] + d2_bhh[Hc + col];
        b2Ni[f] = d2_bih[2 * Hc + col];
        b2Nh[f] = d2_bhh[2 * Hc + col];
    }
    float bO = l2_b[wave * 16 + r];

    float lsum = 0.0f;

    for (int t = 0; t < Sc; ++t) {
        // warm WGP cache for this step's streamed cell2 panels (overlaps cell1)
        {
            int n0 = wave * 32 + r + (half ? Hc : 0);
            __builtin_prefetch(d2Wih + (size_t)n0 * HHc, 0, 0);
            __builtin_prefetch(d2Whh + (size_t)n0 * Hc, 0, 0);
            __builtin_prefetch(d2Whh + (size_t)(2 * Hc + n0) * Hc, 0, 0);
        }
        // ================= cell1: h1 = GRU(gi1, h1) ========================
        {
            v8f aR, aZ, aNh;
#pragma unroll
            for (int j = 0; j < 8; ++j) {
                int m = half * 8 + j;
                aR[j]  = gi1[m * DEC_LDG + c1] + b1R;
                aZ[j]  = gi1[m * DEC_LDG + HHc + c1] + b1Z;
                aNh[j] = b1N;
            }
#pragma unroll
            for (int kb = 0; kb < HHc; kb += 32) {
                AF a = load_a_ptr(&h1_bf[r * DEC_LDH1 + 8 * half], kb);
                aR  = wmma_bf16(a, load_b(w1L, wave * 16,           HHc, kb, r, half), aR);
                aZ  = wmma_bf16(a, load_b(w1L, HHc + wave * 16,     HHc, kb, r, half), aZ);
                aNh = wmma_bf16(a, load_b(w1L, 2 * HHc + wave * 16, HHc, kb, r, half), aNh);
            }
            __syncthreads();
#pragma unroll
            for (int j = 0; j < 8; ++j) {
                int m = half * 8 + j;
                float rg = sigmoidf(aR[j]);
                float zg = sigmoidf(aZ[j]);
                float ng = tanhf(gi1[m * DEC_LDG + 2 * HHc + c1] + rg * aNh[j]);
                float ho = h1_f[m * DEC_LDH1 + c1];
                float hn = (1.0f - zg) * ng + zg * ho;
                h1_f[m * DEC_LDH1 + c1] = hn;
                h1_bf[m * DEC_LDH1 + c1] = f2bf(hn);
            }
            __syncthreads();
        }
        // ========== cell2: gi2 = h1 @ d2_Wih^T fused; h2 = GRU(gi2,h2) =====
        {
            v8f aR[2], aZ[2], aNi[2], aNh[2];
#pragma unroll
            for (int f = 0; f < 2; ++f)
#pragma unroll
                for (int j = 0; j < 8; ++j) {
                    aR[f][j] = b2R[f]; aZ[f][j] = b2Z[f];
                    aNi[f][j] = b2Ni[f]; aNh[f][j] = b2Nh[f];
                }
#pragma unroll
            for (int kb = 0; kb < HHc; kb += 32) {       // input part (K=HH)
                AF a = load_a_ptr(&h1_bf[r * DEC_LDH1 + 8 * half], kb);
#pragma unroll
                for (int f = 0; f < 2; ++f) {
                    int n0 = wave * 32 + f * 16;
                    aR[f]  = wmma_bf16(a, load_b(d2Wih, n0,          HHc, kb, r, half), aR[f]);
                    aZ[f]  = wmma_bf16(a, load_b(d2Wih, Hc + n0,     HHc, kb, r, half), aZ[f]);
                    aNi[f] = wmma_bf16(a, load_b(d2Wih, 2 * Hc + n0, HHc, kb, r, half), aNi[f]);
                }
            }
#pragma unroll
            for (int kb = 0; kb < Hc; kb += 32) {        // hidden part (K=H)
                AF a = load_a_ptr(&h2_bf[r * DEC_LDH2 + 8 * half], kb);
#pragma unroll
                for (int f = 0; f < 2; ++f) {
                    int n0 = wave * 32 + f * 16;
                    aR[f]  = wmma_bf16(a, load_b(d2Whh, n0,          Hc, kb, r, half), aR[f]);
                    aZ[f]  = wmma_bf16(a, load_b(d2Whh, Hc + n0,     Hc, kb, r, half), aZ[f]);
                    aNh[f] = wmma_bf16(a, load_b(d2Whh, 2 * Hc + n0, Hc, kb, r, half), aNh[f]);
                }
            }
            __syncthreads();
#pragma unroll
            for (int f = 0; f < 2; ++f) {
                int col = wave * 32 + f * 16 + r;
#pragma unroll
                for (int j = 0; j < 8; ++j) {
                    int m = half * 8 + j;
                    float rg = sigmoidf(aR[f][j]);
                    float zg = sigmoidf(aZ[f][j]);
                    float ng = tanhf(aNi[f][j] + rg * aNh[f][j]);
                    float ho = h2_f[m * DEC_LDH2 + col];
                    float hn = (1.0f - zg) * ng + zg * ho;
                    h2_f[m * DEC_LDH2 + col] = hn;
                    h2_bf[m * DEC_LDH2 + col] = f2bf(hn);
                }
            }
            __syncthreads();
        }
        // ============ out = h2 @ l2_W^T + b; accumulate MSE ================
        {
            v8f aO;
#pragma unroll
            for (int j = 0; j < 8; ++j) aO[j] = bO;
#pragma unroll
            for (int kb = 0; kb < Hc; kb += 32) {
                AF a = load_a_ptr(&h2_bf[r * DEC_LDH2 + 8 * half], kb);
                aO = wmma_bf16(a, load_b(w2L, wave * 16, Hc, kb, r, half), aO);
            }
#pragma unroll
            for (int j = 0; j < 8; ++j) {
                int m = half * 8 + j;
                int col = wave * 16 + r;
                float d = aO[j] - emb[((size_t)(b0 + m) * Sc + t) * Dc + col];
                lsum += d * d;
            }
            __syncthreads();
        }
    }

    for (int off = 16; off > 0; off >>= 1)
        lsum += __shfl_down(lsum, off, 32);
    if (lane == 0) redbuf[wave] = lsum;
    __syncthreads();
    if (tid == 0) {
        float s = 0.0f;
#pragma unroll
        for (int w = 0; w < 8; ++w) s += redbuf[w];
        atomicAdd(loss_out, s * (1.0f / 16777216.0f));   // mean over B*S*D
    }
}

// ---------------------------------------------------------------------------
extern "C" void kernel_launch(void* const* d_in, const int* in_sizes, int n_in,
                              void* d_out, int out_size, void* d_ws, size_t ws_size,
                              hipStream_t stream) {
    const int*   tokens  = (const int*)  d_in[0];
    const float* path_w  = (const float*)d_in[1];
    const float* sys_w   = (const float*)d_in[2];
    const float* e1_Wih  = (const float*)d_in[3];
    const float* e1_Whh  = (const float*)d_in[4];
    const float* e1_bih  = (const float*)d_in[5];
    const float* e1_bhh  = (const float*)d_in[6];
    const float* e2_Wih  = (const float*)d_in[7];
    const float* e2_Whh  = (const float*)d_in[8];
    const float* e2_bih  = (const float*)d_in[9];
    const float* e2_bhh  = (const float*)d_in[10];
    const float* l1_W    = (const float*)d_in[11];
    const float* l1_b    = (const float*)d_in[12];
    const float* d1_Wih  = (const float*)d_in[13];
    const float* d1_Whh  = (const float*)d_in[14];
    const float* d1_bih  = (const float*)d_in[15];
    const float* d1_bhh  = (const float*)d_in[16];
    const float* d2_Wih  = (const float*)d_in[17];
    const float* d2_Whh  = (const float*)d_in[18];
    const float* d2_bih  = (const float*)d_in[19];
    const float* d2_bhh  = (const float*)d_in[20];
    const float* l2_W    = (const float*)d_in[21];
    const float* l2_b    = (const float*)d_in[22];

    float* out_latent = (float*)d_out;            // [B, L]
    float* out_loss   = (float*)d_out + Bc * Lc;  // scalar

    char* base = (char*)d_ws;
    size_t off = 0;
    auto alloc = [&](size_t bytes) -> void* {
        off = (off + 255) & ~(size_t)255;
        void* p = base + off;
        off += bytes;
        return p;
    };
    float* emb_f    = (float*)alloc((size_t)Bc * Sc * Dc * 4);
    u16*   emb_bf   = (u16*)  alloc((size_t)Bc * Sc * Dc * 2);
    u16*   h1_seq   = (u16*)  alloc((size_t)Bc * Sc * Hc * 2);
    u16*   h2_seq   = (u16*)  alloc((size_t)Bc * Sc * HHc * 2);
    u16*   latentbf = (u16*)  alloc((size_t)Bc * Lc * 2);
    u16*   w_e1Wih  = (u16*)  alloc((size_t)3 * Hc  * Dc  * 2);
    u16*   w_e1Whh  = (u16*)  alloc((size_t)3 * Hc  * Hc  * 2);
    u16*   w_e2Wih  = (u16*)  alloc((size_t)3 * HHc * Hc  * 2);
    u16*   w_e2Whh  = (u16*)  alloc((size_t)3 * HHc * HHc * 2);
    u16*   w_l1W    = (u16*)  alloc((size_t)Lc * HHc * 2);
    u16*   w_d1Wih  = (u16*)  alloc((size_t)3 * HHc * Lc  * 2);
    u16*   w_d1Whh  = (u16*)  alloc((size_t)3 * HHc * HHc * 2);
    u16*   w_d2Wih  = (u16*)  alloc((size_t)3 * Hc  * HHc * 2);
    u16*   w_d2Whh  = (u16*)  alloc((size_t)3 * Hc  * Hc  * 2);
    u16*   w_l2W    = (u16*)  alloc((size_t)Dc * Hc * 2);

    zero_loss_kernel<<<1, 1, 0, stream>>>(out_loss);

    auto conv = [&](const float* s, u16* d, int n) {
        convert_bf16_kernel<<<(n + 255) / 256, 256, 0, stream>>>(s, d, n);
    };
    conv(e1_Wih, w_e1Wih, 3 * Hc * Dc);
    conv(e1_Whh, w_e1Whh, 3 * Hc * Hc);
    conv(e2_Wih, w_e2Wih, 3 * HHc * Hc);
    conv(e2_Whh, w_e2Whh, 3 * HHc * HHc);
    conv(l1_W,   w_l1W,   Lc * HHc);
    conv(d1_Wih, w_d1Wih, 3 * HHc * Lc);
    conv(d1_Whh, w_d1Whh, 3 * HHc * HHc);
    conv(d2_Wih, w_d2Wih, 3 * Hc * HHc);
    conv(d2_Whh, w_d2Whh, 3 * Hc * Hc);
    conv(l2_W,   w_l2W,   Dc * Hc);

    embed_kernel<<<(Bc * Sc * Dc) / 256, 256, 0, stream>>>(
        tokens, path_w, sys_w, emb_f, emb_bf);

    // layer1: stage Wih (192KB) in LDS; Whh (384KB) streams from L2.
    gru_layer_kernel<Hc, Dc, true, false>
        <<<Bc / 16, 256, gru_smem_bytes(Hc, Dc, true, false), stream>>>(
            emb_bf, w_e1Wih, w_e1Whh, e1_bih, e1_bhh, h1_seq);

    // layer2: stage both Wih (192KB) and Whh (96KB); total ~301KB < 320KB WGP.
    gru_layer_kernel<HHc, Hc, true, true>
        <<<Bc / 16, 256, gru_smem_bytes(HHc, Hc, true, true), stream>>>(
            h1_seq, w_e2Wih, w_e2Whh, e2_bih, e2_bhh, h2_seq);

    latent_kernel<<<Bc / 16, 128, 0, stream>>>(
        h2_seq, w_l1W, l1_b, out_latent, latentbf);

    decoder_kernel<<<Bc / 16, 256, DEC_SMEM, stream>>>(
        latentbf,
        w_d1Wih, w_d1Whh, d1_bih, d1_bhh,
        w_d2Wih, w_d2Whh, d2_bih, d2_bhh,
        w_l2W, l2_b, emb_f, out_loss);
}
